// MultiHeadAttention_88888643158482
// MI455X (gfx1250) — matrix operand order repaired and checked by reading, not compiled, not run on hardware
//
#include <hip/hip_runtime.h>
#include <hip/hip_bf16.h>
#include <stdint.h>

// MI455X / gfx1250 multi-head attention, bf16 WMMA + async LDS staging.
// B=4, N=2048, C=1024, H=16, D=64.

typedef __attribute__((ext_vector_type(16))) __bf16 v16bf;
typedef __attribute__((ext_vector_type(8)))  float  v8f;

#define BQ 4
#define NQ 2048
#define CQ 1024
#define HQ 16
#define DQ 64
#define BH (BQ * HQ)          // 64

__device__ __forceinline__ v8f wmma_bf16(v16bf a, v16bf b, v8f c) {
  // D = A(16x32) * B(32x16) + C, f32 accumulate
  return __builtin_amdgcn_wmma_f32_16x16x32_bf16(
      /*neg_a=*/false, a, /*neg_b=*/false, b,
      /*c_mod=*/(short)0, c, /*reuse_a=*/false, /*reuse_b=*/false);
}

union FragU { uint4 u[2]; v16bf f; };

// A-operand fragment (16 rows x 32 K) from row-major bf16 (ld = row stride, elems).
// lane holds row (lane&15); K elems = 16*(lane>=16) + [0..15].
__device__ __forceinline__ v16bf load_a(const __bf16* p, int ld, int lane) {
  const int r = lane & 15;
  const int kh = (lane >> 4) << 4;
  const __bf16* q = p + r * ld + kh;
  FragU x;
  x.u[0] = *(const uint4*)(q);
  x.u[1] = *(const uint4*)(q + 8);
  return x.f;
}

// B-operand fragment (32 K x 16 cols) from row-major "B^T" bf16 (row = out col,
// contiguous K). lane holds col (lane&15); K elems = 16*(lane>=16) + [0..15].
__device__ __forceinline__ v16bf load_b(const __bf16* p, int ld, int lane) {
  const int n = lane & 15;
  const int kh = (lane >> 4) << 4;
  const __bf16* q = p + n * ld + kh;
  FragU x;
  x.u[0] = *(const uint4*)(q);
  x.u[1] = *(const uint4*)(q + 8);
  return x.f;
}

// CDNA5 async global->LDS copy (16B), tracked by ASYNCcnt. GV addressing.
__device__ __forceinline__ void async_copy_b128(__bf16* lds_dst, const __bf16* gsrc) {
  const uint32_t l = (uint32_t)(uintptr_t)lds_dst;   // low 32 bits = LDS offset
  asm volatile("global_load_async_to_lds_b128 %0, %1, off"
               :: "v"(l), "v"(gsrc) : "memory");
}
__device__ __forceinline__ void wait_async0() {
  asm volatile("s_wait_asynccnt 0" ::: "memory");
}

// ---------------------------------------------------------------------------
// Kernel 1: qkv = x @ W^T + b  (M=8192, Nout=3072, K=1024), bf16 WMMA, f32 acc.
// LDS-staged: per K-step the block converts a 128x32 x-tile and 128x32 W-tile
// to bf16 in LDS (double buffered, 1 barrier/step), then 8 waves build
// fragments from LDS. Writes Q [BH,N,D] (x 1/sqrt(D)), K [BH,N,D], V^T [BH,D,N].
// ---------------------------------------------------------------------------
__global__ __launch_bounds__(256) void qkv_gemm_kernel(
    const float* __restrict__ x, const float* __restrict__ w,
    const float* __restrict__ bias,
    __bf16* __restrict__ qb, __bf16* __restrict__ kb, __bf16* __restrict__ vt) {
  __shared__ __align__(16) __bf16 As[2][128 * 32];
  __shared__ __align__(16) __bf16 Bs[2][128 * 32];

  const int tid = threadIdx.x;
  const int lane = tid & 31;
  const int wave = tid >> 5;
  const int wm = wave & 3;                 // 4 waves along M
  const int wn = wave >> 2;                // 2 waves along N
  const int mblk = blockIdx.y * 128;
  const int oblk = blockIdx.x * 128;

  // Cooperative stage: thread -> one 16-float chunk of x-tile and of W-tile.
  auto stage = [&](int buf, int k0) {
    const int row = tid >> 1;              // 0..127
    const int half = (tid & 1) << 4;       // 0 or 16
    const float* ax = x + (size_t)(mblk + row) * CQ + k0 + half;
    const float* bw = w + (size_t)(oblk + row) * CQ + k0 + half;
    __bf16* ad = &As[buf][row * 32 + half];
    __bf16* bd = &Bs[buf][row * 32 + half];
#pragma unroll
    for (int i = 0; i < 16; ++i) {
      ad[i] = (__bf16)ax[i];
      bd[i] = (__bf16)bw[i];
    }
  };

  v8f acc[2][4];
#pragma unroll
  for (int ns = 0; ns < 4; ++ns) {
    const float bv = bias[oblk + wn * 64 + ns * 16 + (lane & 15)];
#pragma unroll
    for (int ms = 0; ms < 2; ++ms)
#pragma unroll
      for (int v = 0; v < 8; ++v) acc[ms][ns][v] = bv;
  }

  stage(0, 0);
  for (int it = 0; it < CQ / 32; ++it) {
    __syncthreads();
    const int cur = it & 1;
    if (it + 1 < CQ / 32) stage(cur ^ 1, (it + 1) * 32);

    v16bf a[2], bf[4];
#pragma unroll
    for (int ms = 0; ms < 2; ++ms)
      a[ms] = load_a(&As[cur][(wm * 32 + ms * 16) * 32], 32, lane);
#pragma unroll
    for (int ns = 0; ns < 4; ++ns)
      bf[ns] = load_b(&Bs[cur][(wn * 64 + ns * 16) * 32], 32, lane);
#pragma unroll
    for (int ms = 0; ms < 2; ++ms)
#pragma unroll
      for (int ns = 0; ns < 4; ++ns)
        acc[ms][ns] = wmma_bf16(a[ms], bf[ns], acc[ms][ns]);
  }

  // Branchless scatter epilogue: every lane stores exactly one bf16 per value.
  const int hl = lane >> 4;
#pragma unroll
  for (int ms = 0; ms < 2; ++ms) {
#pragma unroll
    for (int ns = 0; ns < 4; ++ns) {
      const int o = oblk + wn * 64 + ns * 16 + (lane & 15);
      const int which = o >> 10;                 // 0:Q 1:K 2:V
      const int rem = o & (CQ - 1);
      const int h = rem >> 6;
      const int d = rem & (DQ - 1);
      const float scale = (which == 0) ? 0.125f : 1.0f;
      __bf16* base = (which == 0) ? qb : ((which == 1) ? kb : vt);
#pragma unroll
      for (int v = 0; v < 8; ++v) {
        const int m = mblk + wm * 32 + ms * 16 + hl * 8 + v;
        const int b = m >> 11;
        const int n = m & (NQ - 1);
        const int bh = b * HQ + h;
        const size_t offQK = (size_t)(bh * NQ + n) * DQ + d;
        const size_t offV  = (size_t)(bh * DQ + d) * NQ + n;
        base[(which == 2) ? offV : offQK] = (__bf16)(acc[ms][ns][v] * scale);
      }
    }
  }
}

// ---------------------------------------------------------------------------
// Kernel 2: flash attention. Block = 8 waves = 128 queries of one (b,h).
// Per 32-key step, the block async-DMAs the K block (32x64) and V^T block
// (64x32) into LDS once (double buffered, ASYNCcnt + 1 barrier/step); each
// wave builds fragments from LDS. S^T = K*Q^T, online softmax per query
// column, O^T += V^T * P^T.
// ---------------------------------------------------------------------------
__global__ __launch_bounds__(256) void attn_kernel(
    const __bf16* __restrict__ qb, const __bf16* __restrict__ kb,
    const __bf16* __restrict__ vt, float* __restrict__ out) {
  __shared__ __align__(16) __bf16 Ks[2][32 * 64];   // [key][d]
  __shared__ __align__(16) __bf16 Vs[2][64 * 32];   // [d][key]

  const int tid = threadIdx.x;
  const int lane = tid & 31;
  const int wave = tid >> 5;
  const int bh = blockIdx.y;
  const int b = bh >> 4, h = bh & 15;
  const int q0 = blockIdx.x * 128 + wave * 16;

  const __bf16* Q = qb + (size_t)bh * NQ * DQ;
  const __bf16* K = kb + (size_t)bh * NQ * DQ;
  const __bf16* V = vt + (size_t)bh * DQ * NQ;      // [D][N]

  // Async stage of one 32-key block: K rows are contiguous (4KB), V^T strided.
  auto stage = [&](int buf, int j0) {
    async_copy_b128(&Ks[buf][tid * 8], K + (size_t)j0 * DQ + tid * 8);
    const int d = tid >> 2;                // 0..63
    const int c = (tid & 3) << 3;          // 0,8,16,24
    async_copy_b128(&Vs[buf][d * 32 + c], V + (size_t)d * NQ + j0 + c);
  };

  // Q^T B-fragments, persistent (d chunks 0..31, 32..63)
  v16bf qf[2];
#pragma unroll
  for (int c = 0; c < 2; ++c)
    qf[c] = load_b(Q + (size_t)q0 * DQ + c * 32, DQ, lane);

  v8f oacc[4];
#pragma unroll
  for (int t = 0; t < 4; ++t)
#pragma unroll
    for (int v = 0; v < 8; ++v) oacc[t][v] = 0.0f;
  float mrun = -3.0e38f, lrun = 0.0f;
  const bool hiHalf = (lane >> 4) != 0;

  stage(0, 0);
  for (int it = 0; it < NQ / 32; ++it) {
    const int cur = it & 1;
    wait_async0();                         // my async stores for buf 'cur' done
    __syncthreads();                       // everyone's staging visible
    if (it + 1 < NQ / 32) stage(cur ^ 1, (it + 1) * 32);

    // S^T tiles: tile0 = keys 0..15, tile1 = keys 16..31 of this block.
    v8f s0 = {}, s1 = {};
#pragma unroll
    for (int c = 0; c < 2; ++c) {
      v16bf ak0 = load_a(&Ks[cur][0 * 64 + c * 32], 64, lane);
      v16bf ak1 = load_a(&Ks[cur][16 * 64 + c * 32], 64, lane);
      s0 = wmma_bf16(ak0, qf[c], s0);
      s1 = wmma_bf16(ak1, qf[c], s1);
    }

    // --- online softmax over the 32 keys (per query column) ---
    float mloc = s0[0];
#pragma unroll
    for (int v = 0; v < 8; ++v) {
      mloc = fmaxf(mloc, s0[v]);
      mloc = fmaxf(mloc, s1[v]);
    }
    mloc = fmaxf(mloc, __shfl_xor(mloc, 16, 32));
    const float mnew = fmaxf(mrun, mloc);
    const float r = __expf(mrun - mnew);
    float p0[8], p1[8], sloc = 0.0f;
#pragma unroll
    for (int v = 0; v < 8; ++v) {
      p0[v] = __expf(s0[v] - mnew);
      p1[v] = __expf(s1[v] - mnew);
      sloc += p0[v] + p1[v];
    }
    sloc += __shfl_xor(sloc, 16, 32);
    lrun = lrun * r + sloc;
    mrun = mnew;
#pragma unroll
    for (int t = 0; t < 4; ++t)
#pragma unroll
      for (int v = 0; v < 8; ++v) oacc[t][v] *= r;

    // Pack P^T into B-fragment: lane needs keys 16*half+[0..15]; C-layout
    // gives keys 8*half+v per tile -> one half-wave shuffle fixes it up.
    v16bf pb;
#pragma unroll
    for (int v = 0; v < 8; ++v) {
      const float x0 = __shfl_xor(p0[v], 16, 32);
      const float x1 = __shfl_xor(p1[v], 16, 32);
      pb[v]     = (__bf16)(hiHalf ? x1 : p0[v]);   // K = 16*half + v
      pb[v + 8] = (__bf16)(hiHalf ? p1[v] : x0);   // K = 16*half + 8 + v
    }

    // O^T(d x 16q) += V^T(d x 32keys) * P^T(32keys x 16q); 4 d-tiles of 16.
#pragma unroll
    for (int t = 0; t < 4; ++t) {
      v16bf av = load_a(&Vs[cur][(t * 16) * 32], 32, lane);
      oacc[t] = wmma_bf16(av, pb, oacc[t]);
    }
  }

  // Epilogue: normalize and write out[b, q, h*64 + d] (f32).
  const float inv = 1.0f / lrun;
  const int qn = q0 + (lane & 15);
  const int hl = lane >> 4;
#pragma unroll
  for (int t = 0; t < 4; ++t)
#pragma unroll
    for (int v = 0; v < 8; ++v) {
      const int d = t * 16 + hl * 8 + v;
      out[(size_t)(b * NQ + qn) * CQ + h * DQ + d] = oacc[t][v] * inv;
    }
}

// ---------------------------------------------------------------------------
extern "C" void kernel_launch(void* const* d_in, const int* in_sizes, int n_in,
                              void* d_out, int out_size, void* d_ws, size_t ws_size,
                              hipStream_t stream) {
  const float* x    = (const float*)d_in[0];   // [B,N,C]
  const float* w    = (const float*)d_in[1];   // [3C,C]
  const float* bias = (const float*)d_in[2];   // [3C]
  float* out = (float*)d_out;                  // [B,N,C] f32

  __bf16* ws = (__bf16*)d_ws;
  const size_t per = (size_t)BH * NQ * DQ;     // 8,388,608 elems (16 MB) each
  __bf16* qb = ws;                             // Q  [BH,N,D], pre-scaled
  __bf16* kb = ws + per;                       // K  [BH,N,D]
  __bf16* vt = ws + 2 * per;                   // V^T[BH,D,N]

  dim3 gGemm(3 * CQ / 128, (BQ * NQ) / 128);   // 24 x 64
  qkv_gemm_kernel<<<gGemm, dim3(256), 0, stream>>>(x, w, bias, qb, kb, vt);

  dim3 gAttn(NQ / 128, BH);                    // 16 x 64
  attn_kernel<<<gAttn, dim3(256), 0, stream>>>(qb, kb, vt, out);
}
